// MultiDistanceChannelClustering_4174708212241
// MI455X (gfx1250) — matrix operand order repaired and checked by reading, not compiled, not run on hardware
//
#include <hip/hip_runtime.h>
#include <hip/hip_bf16.h>
#include <math.h>

// ---------------------------------------------------------------------------
// MultiDistanceChannelClustering for MI455X (gfx1250)
//   X[16,64,2048] -> |rfft| -> pairwise channel L2 via bf16 WMMA Gram matrix
//   -> RBF / row-max / gumbel-sigmoid / hard threshold -> out[16,1,64,64]
// ---------------------------------------------------------------------------

typedef __bf16 bf16;
typedef __attribute__((ext_vector_type(8)))  bf16  bf16x8;
typedef __attribute__((ext_vector_type(16))) bf16  v16bf;
typedef __attribute__((ext_vector_type(8)))  float v8f;

#define B_DIM   16
#define C_DIM   64
#define T_DIM   2048
#define LOG2T   11
#define F_DIM   1025          // T/2 + 1
#define F_PAD   1056          // next multiple of 32 (WMMA K-step), zero padded
#define INV_2SIG2 (1.0f / 4.5f)   // 1/(2*1.5^2)
#define THRESH  0.7f

// ---------------------------------------------------------------------------
// Kernel 1: per-row 2048-pt Stockham radix-2 FFT in LDS, |XF| -> bf16 (padded),
//           exact f32 row norm (sum of |XF|^2).
// grid = B*C blocks, 256 threads each. LDS: 2 ping-pong complex buffers = 32KB.
// ---------------------------------------------------------------------------
__global__ __launch_bounds__(256)
void fft_mag_kernel(const float* __restrict__ X,
                    bf16* __restrict__ XFb,
                    float* __restrict__ norms) {
  __shared__ float sre[2][T_DIM];
  __shared__ float sim[2][T_DIM];
  __shared__ float sred[256];

  const int row = blockIdx.x;            // b*C + c
  const int tid = threadIdx.x;
  const float* xr = X + (size_t)row * T_DIM;

  for (int t = tid; t < T_DIM; t += 256) { sre[0][t] = xr[t]; sim[0][t] = 0.0f; }
  __syncthreads();

  int cur = 0;
  #pragma unroll 1
  for (int st = 0; st < LOG2T; ++st) {
    const int   s      = 1 << st;
    const int   ncur   = T_DIM >> st;
    const int   m      = ncur >> 1;
    const float theta0 = -6.28318530717958647692f / (float)ncur;
    const int   nxt    = cur ^ 1;
    for (int u = tid; u < (T_DIM / 2); u += 256) {
      const int p = u >> st;             // u / s
      const int q = u & (s - 1);         // u % s
      float ws, wc;
      __sincosf(theta0 * (float)p, &ws, &wc);   // w = exp(-i*2pi*p/ncur)
      const int ia = q + s * p;
      const int ib = ia + s * m;
      const float ar = sre[cur][ia], ai = sim[cur][ia];
      const float br = sre[cur][ib], bi = sim[cur][ib];
      const int iy0 = q + s * (2 * p);
      const int iy1 = iy0 + s;
      sre[nxt][iy0] = ar + br;
      sim[nxt][iy0] = ai + bi;
      const float dr = ar - br, di = ai - bi;
      sre[nxt][iy1] = dr * wc - di * ws;
      sim[nxt][iy1] = dr * ws + di * wc;
    }
    __syncthreads();
    cur = nxt;
  }

  // magnitudes (bf16, zero-padded to F_PAD) + f32 norm reduction
  float acc = 0.0f;
  bf16* orow = XFb + (size_t)row * F_PAD;
  for (int f = tid; f < F_PAD; f += 256) {
    float mag = 0.0f;
    if (f < F_DIM) {
      const float re = sre[cur][f], im = sim[cur][f];
      const float m2 = re * re + im * im;
      acc += m2;
      mag = sqrtf(m2);
    }
    orow[f] = (bf16)mag;
  }
  sred[tid] = acc;
  __syncthreads();
  #pragma unroll
  for (int o = 128; o > 0; o >>= 1) {
    if (tid < o) sred[tid] += sred[tid + o];
    __syncthreads();
  }
  if (tid == 0) norms[row] = sred[0];
}

// ---------------------------------------------------------------------------
// Kernel 2: batched Gram matrix G[b] = XF[b] * XF[b]^T  (64x64, K = 1056)
// One wave32 per 16x16 tile: grid = 16 batches * 4*4 tiles = 256 blocks of 32.
// Fragment layout per CDNA5 ISA (16-bit A 16x32 / B 32x16 tables):
//   A lane L: m = L&15, h = L>>4; elems 0..7 = K[k0+h*8 ..], 8..15 = K[k0+16+h*8 ..]
//   B lane L: n = L&15, h = L>>4; elems 0..7 = K[k0+h*8 ..], 8..15 = K[k0+16+h*8 ..]
// ---------------------------------------------------------------------------
__global__ __launch_bounds__(32)
void gram_wmma_kernel(const bf16* __restrict__ XFb, float* __restrict__ gram) {
  const int b    = blockIdx.x >> 4;
  const int tm   = (blockIdx.x >> 2) & 3;
  const int tn   = blockIdx.x & 3;
  const int lane = threadIdx.x & 31;
  const int half = lane >> 4;
  const int l16  = lane & 15;

  const bf16* Abase = XFb + ((size_t)(b * C_DIM + tm * 16 + l16)) * F_PAD + half * 8;
  const bf16* Bbase = XFb + ((size_t)(b * C_DIM + tn * 16 + l16)) * F_PAD + half * 8;

  v8f acc = {};
  #pragma unroll 4
  for (int k0 = 0; k0 < F_PAD; k0 += 32) {
    const bf16x8 a0 = *(const bf16x8*)(Abase + k0);        // global_load_b128
    const bf16x8 a1 = *(const bf16x8*)(Abase + k0 + 16);
    const bf16x8 b0 = *(const bf16x8*)(Bbase + k0);
    const bf16x8 b1 = *(const bf16x8*)(Bbase + k0 + 16);
    const v16bf av = __builtin_shufflevector(a0, a1, 0,1,2,3,4,5,6,7,8,9,10,11,12,13,14,15);
    const v16bf bv = __builtin_shufflevector(b0, b1, 0,1,2,3,4,5,6,7,8,9,10,11,12,13,14,15);
    // D = A*B + C : v_wmma_f32_16x16x32_bf16
    acc = __builtin_amdgcn_wmma_f32_16x16x32_bf16(
        /*neg_a=*/false, av, /*neg_b=*/false, bv,
        /*c_mod=*/(short)0, acc, /*reuse_a=*/false, /*reuse_b=*/false);
  }

  // D layout: VGPR v -> (lanes 0-15: M=v, lanes 16-31: M=8+v), N = lane&15
  float* g = gram + (size_t)b * (C_DIM * C_DIM);
  const int nCol = tn * 16 + l16;
  #pragma unroll
  for (int v = 0; v < 8; ++v) {
    const int mRow = tm * 16 + half * 8 + v;
    g[mRow * C_DIM + nCol] = acc[v];
  }
}

// ---------------------------------------------------------------------------
// Kernel 3: per-(b,i) row epilogue. euclid = n_i + n_j - 2*G_ij, RBF,
// reciprocal, row-max via wave32 shuffle, gumbel-sigmoid, straight-through.
// grid = B*C blocks of one wave32; each lane handles j = lane and lane+32.
// ---------------------------------------------------------------------------
__global__ __launch_bounds__(32)
void finalize_kernel(const float* __restrict__ gram,
                     const float* __restrict__ norms,
                     const float* __restrict__ gumbel,
                     float* __restrict__ out) {
  const int b    = blockIdx.x >> 6;     // / 64
  const int i    = blockIdx.x & 63;
  const int lane = threadIdx.x;

  const float  ni   = norms[b * C_DIM + i];
  const float* grow = gram  + ((size_t)b * C_DIM + i) * C_DIM;
  const float* nrow = norms + (size_t)b * C_DIM;

  float ed[2];
  #pragma unroll
  for (int t = 0; t < 2; ++t) {
    const int j = lane + t * 32;
    float e = ni + nrow[j] - 2.0f * grow[j];
    e = fmaxf(e, 0.0f);                       // guard bf16 rounding
    const float dist = 1.0f - expf(-e * INV_2SIG2);
    const float v = 1.0f / (dist + 1e-10f);
    ed[t] = (j == i) ? 0.0f : v;              // zero diagonal before row max
  }

  float mx = fmaxf(ed[0], ed[1]);
  #pragma unroll
  for (int o = 16; o > 0; o >>= 1)            // wave32 reduction (gfx1250)
    mx = fmaxf(mx, __shfl_xor(mx, o, 32));

  const float* gmb  = gumbel + ((size_t)b * C_DIM + i) * C_DIM;
  float*       orow = out    + ((size_t)b * C_DIM + i) * C_DIM;
  #pragma unroll
  for (int t = 0; t < 2; ++t) {
    const int j = lane + t * 32;
    const float p     = (j == i) ? 0.99f : (ed[t] / mx) * 0.99f;
    const float logit = logf(p + 1e-10f) - logf(1.0f - p + 1e-10f) + gmb[j];
    const float s     = 1.0f / (1.0f + expf(-logit));
    const float hard  = (s > THRESH) ? 1.0f : 0.0f;
    orow[j] = (hard - s) + s;                 // straight-through forward value
  }
}

// ---------------------------------------------------------------------------
extern "C" void kernel_launch(void* const* d_in, const int* in_sizes, int n_in,
                              void* d_out, int out_size, void* d_ws, size_t ws_size,
                              hipStream_t stream) {
  const float* X      = (const float*)d_in[0];   // [16,64,2048] f32
  const float* gumbel = (const float*)d_in[1];   // [16,64,64]   f32
  float*       out    = (float*)d_out;           // [16,1,64,64] f32

  char* ws = (char*)d_ws;
  const size_t nRows = (size_t)B_DIM * C_DIM;    // 1024

  bf16* XFb = (bf16*)ws;                               // 1024*1056*2  ~2.1 MB
  size_t off = nRows * F_PAD * sizeof(bf16);
  off = (off + 255) & ~(size_t)255;
  float* norms = (float*)(ws + off);                   // 1024 f32
  off += nRows * sizeof(float);
  off = (off + 255) & ~(size_t)255;
  float* gram = (float*)(ws + off);                    // 16*64*64 f32 = 256 KB
  (void)ws_size; (void)in_sizes; (void)n_in; (void)out_size;

  fft_mag_kernel  <<<dim3((unsigned)nRows), dim3(256), 0, stream>>>(X, XFb, norms);
  gram_wmma_kernel<<<dim3(B_DIM * 16),      dim3(32),  0, stream>>>(XFb, gram);
  finalize_kernel <<<dim3((unsigned)nRows), dim3(32),  0, stream>>>(gram, norms, gumbel, out);
}
